// Decoder_78864189489886
// MI455X (gfx1250) — matrix operand order repaired and checked by reading, not compiled, not run on hardware
//
#include <hip/hip_runtime.h>
#include <math.h>

// ---------------------------------------------------------------------------
// Types for CDNA5 (gfx1250) WMMA: v_wmma_f32_16x16x32_bf16
// ---------------------------------------------------------------------------
typedef __attribute__((ext_vector_type(8)))  __bf16 v8bf;
typedef __attribute__((ext_vector_type(16))) __bf16 v16bf;
typedef __attribute__((ext_vector_type(8)))  float  v8f;

// D = A(16x32 bf16) x B(32x16 bf16) + C(16x16 f32)
__device__ __forceinline__ v8f wmma_bf16(v16bf a, v16bf b, v8f c) {
    return __builtin_amdgcn_wmma_f32_16x16x32_bf16(
        /*neg_a=*/false, a, /*neg_b=*/false, b,
        /*c_mod=*/(short)0, c, /*reuse_a=*/false, /*reuse_b=*/false);
}

// Load a 16-element bf16 fragment as two 16-byte chunks at (base+off) and
// (base+off+gap). `off` is a per-lane 32-bit byte offset; all loop deltas are
// compile-time constants so they fold into the instruction IOFFSET field
// (SGPR base + 32-bit VGPR offset + imm addressing).
__device__ __forceinline__ v16bf ld16(const __bf16* base, unsigned off, int gap) {
    const char* p = (const char*)base + off;
    v8bf lo = *(const v8bf*)p;
    v8bf hi = *(const v8bf*)(p + gap);
    return __builtin_shufflevector(lo, hi,
        0,1,2,3,4,5,6,7,8,9,10,11,12,13,14,15);
}
// A-fragment (16x32): lane L = row (L&15); 16 contiguous K     -> gap = 16 B.
// B-fragment (32x16): lane L = col (L&15); 16 contiguous K     -> gap = 16 B.

__device__ __forceinline__ float sigmoidf_(float x) {
    return 1.0f / (1.0f + __expf(-x));
}

// ---------------------------------------------------------------------------
// fp32 -> bf16 convert (weights, encoder outputs, fcn latent)
// ---------------------------------------------------------------------------
__global__ void cvt_bf16_kernel(const float* __restrict__ src,
                                __bf16* __restrict__ dst, int n) {
    int i = blockIdx.x * blockDim.x + threadIdx.x;
    if (i < n) dst[i] = (__bf16)src[i];
}

// Whh1_fold[i][j] = sum_r Whh1[i][j + 64r]   (because h_rep = tile(h0, 4))
__global__ void fold_whh1_kernel(const float* __restrict__ W,   // 768 x 256
                                 __bf16* __restrict__ out) {    // 768 x 64
    int i = blockIdx.x * blockDim.x + threadIdx.x;
    if (i < 768 * 64) {
        int r = i >> 6, c = i & 63;
        float s = W[r * 256 + c] + W[r * 256 + 64 + c] +
                  W[r * 256 + 128 + c] + W[r * 256 + 192 + c];
        out[i] = (__bf16)s;
    }
}

// ---------------------------------------------------------------------------
// scores[j,k] = sum_b h[b,j] * enc[b,k]  (contraction over BZ=4096),
// then row softmax over k, emit bf16 w (hdim x 64).
// Grid: hdim/16 blocks x 1024 threads. Thread = (jl = tid/64, k = tid%64).
// ---------------------------------------------------------------------------
__global__ void scores_softmax_kernel(const float* __restrict__ h, int ldh,
                                      const float* __restrict__ enc, // +t*64, stride 6400
                                      __bf16* __restrict__ wmat) {
    __shared__ float es[64][64];   // enc chunk (batch x 64)
    __shared__ float hs[64][16];   // h chunk   (batch x 16 cols)
    __shared__ float sc[16][64];
    __shared__ float rmax[16], rsum[16];

    const int tid = threadIdx.x;
    const int jl  = tid >> 6;      // 0..15
    const int k   = tid & 63;      // 0..63
    const int jb  = blockIdx.x * 16;

    float acc = 0.0f;
    for (int b0 = 0; b0 < 4096; b0 += 64) {
        for (int i = tid; i < 64 * 64; i += 1024) {
            int bb = i >> 6, kk = i & 63;
            es[bb][kk] = enc[(size_t)(b0 + bb) * 6400 + kk];
        }
        {
            int bb = tid >> 4, jj = tid & 15;
            hs[bb][jj] = h[(size_t)(b0 + bb) * ldh + jb + jj];
        }
        __syncthreads();
#pragma unroll 8
        for (int bb = 0; bb < 64; ++bb)
            acc = fmaf(hs[bb][jl], es[bb][k], acc);
        __syncthreads();
    }
    sc[jl][k] = acc;
    __syncthreads();
    if (k == 0) {
        float m = sc[jl][0];
        for (int i = 1; i < 64; ++i) m = fmaxf(m, sc[jl][i]);
        float s = 0.0f;
        for (int i = 0; i < 64; ++i) s += __expf(sc[jl][i] - m);
        rmax[jl] = m; rsum[jl] = s;
    }
    __syncthreads();
    float wv = __expf(acc - rmax[jl]) / rsum[jl];
    wmat[(size_t)(jb + jl) * 64 + k] = (__bf16)wv;
}

// ---------------------------------------------------------------------------
// Fused step: ctx -> att -> GRU gates -> h_new -> output projection.
// One block (4 wave32) per 16-row batch tile. Grid = 4096/16 = 256 blocks.
//   FIRST:  ctx 64,  att 64,  Wa=Wa1 (64x128),  Wih1 K=64, Whh1_fold K=64,
//           h_prev = tile(fcn, 4).
//   !FIRST: ctx 256, att 256, Wa=Wa2 (256x512), Wih2/Whh2 K=256.
// ---------------------------------------------------------------------------
template <bool FIRST>
__global__ void step_kernel(const __bf16* __restrict__ enc_t,   // bf16, +t*64, stride 6400
                            const __bf16* __restrict__ wmat,    // (DCTX x 64)
                            const __bf16* __restrict__ Wa,      // (DATT x DWA)
                            const float*  __restrict__ ba,
                            const __bf16* __restrict__ Wih,     // (768 x DATT)
                            const __bf16* __restrict__ Whh,     // (768 x LDH)
                            const float*  __restrict__ bih,
                            const float*  __restrict__ bhh,
                            const __bf16* __restrict__ h_att,   // (4096 x LDH) bf16
                            const __bf16* __restrict__ h_gh,    // (4096 x LDH) bf16
                            const float*  __restrict__ h_prev,  // fp32 (LDH cols)
                            float*        __restrict__ h32_out, // (4096 x 256)
                            __bf16*       __restrict__ hbf_out, // (4096 x 256)
                            const __bf16* __restrict__ WoutW,   // (64 x 256)
                            const float*  __restrict__ bout,
                            float*        __restrict__ outp) {  // +t*64, stride 6400
    constexpr int DCTX = FIRST ? 64 : 256;
    constexpr int DATT = FIRST ? 64 : 256;
    constexpr int DWA  = FIRST ? 128 : 512;
    constexpr int LDH  = FIRST ? 64 : 256;

    __shared__ __align__(16) __bf16 ctx_s[16 * DCTX];
    __shared__ __align__(16) __bf16 att_s[16 * DATT];
    __shared__ __align__(16) __bf16 hnew_s[16 * 256];

    const int lane  = threadIdx.x & 31;
    const int wave  = threadIdx.x >> 5;   // 0..3
    const int rbase = blockIdx.x * 16;
    const int n     = lane & 15;
    const int m0    = (lane >> 4) << 3;   // 0 or 8

    // Per-lane 32-bit byte offsets, computed ONCE. All tile/K/gate deltas at
    // the use sites are compile-time constants -> folded into load IOFFSET.
    const unsigned lkA = (unsigned)(((lane >> 4) << 4) * 2);  // A half-lane K off (bytes)
    const unsigned lkB = (unsigned)(((lane >> 4) << 4) * 2);  // B half-lane K off (bytes)
    const unsigned enc_ao  = (unsigned)((rbase + n) * 6400 * 2) + lkA;
    const unsigned h_ao    = (unsigned)((rbase + n) * LDH  * 2) + lkA;
    const unsigned ctx_ao  = (unsigned)(n * DCTX * 2) + lkA;
    const unsigned att_ao  = (unsigned)(n * DATT * 2) + lkA;
    const unsigned hnew_ao = (unsigned)(n * 256  * 2) + lkA;
    const unsigned wmat_bo = (unsigned)(n * 64  * 2) + lkB;
    const unsigned wa_bo   = (unsigned)((wave * 16 + n) * DWA * 2) + lkB;   // tile at=wave+4i
    const unsigned wih_bo  = (unsigned)((wave * 64 + n) * DATT * 2) + lkB;  // cols [64w,64w+64)
    const unsigned whh_bo  = (unsigned)((wave * 64 + n) * LDH  * 2) + lkB;
    const unsigned wout_bo = (unsigned)((wave * 16 + n) * 256  * 2) + lkB;

    // Warm WGP$/L2 for this wave's weight rows (global_prefetch_b8).
    __builtin_prefetch((const char*)Wih + wih_bo, 0, 3);
    __builtin_prefetch((const char*)Whh + whh_bo, 0, 3);

    // ---- stage 1: ctx = enc_t @ w.T ----------------------------------------
#pragma unroll
    for (int i = 0; i < DCTX / 64; ++i) {           // ct = wave + 4*i
        v8f acc = {};
#pragma unroll
        for (int kt = 0; kt < 2; ++kt) {
            v16bf a = ld16(enc_t, enc_ao + kt * 64, 16);
            v16bf b = ld16(wmat, wmat_bo + (unsigned)(wave * 16 * 64 * 2)
                                         + i * (4 * 16 * 64 * 2) + kt * 64, 16);
            acc = wmma_bf16(a, b, acc);
        }
        const int ct = wave + 4 * i;
#pragma unroll
        for (int v = 0; v < 8; ++v)
            ctx_s[(m0 + v) * DCTX + ct * 16 + n] = (__bf16)acc[v];
    }
    __syncthreads();

    // ---- stage 2: att = tanh(ctx @ WaA.T + h @ WaB.T + ba) ------------------
    for (int i = 0; i < DATT / 64; ++i) {           // at = wave + 4*i
        v8f acc = {};
        const unsigned wbo = wa_bo + i * (unsigned)(4 * 16 * DWA * 2);
#pragma unroll 4
        for (int kt = 0; kt < DCTX / 32; ++kt) {
            v16bf a = ld16(ctx_s, ctx_ao + kt * 64, 16);
            v16bf b = ld16(Wa, wbo + kt * 64, 16);
            acc = wmma_bf16(a, b, acc);
        }
#pragma unroll 4
        for (int kt = 0; kt < LDH / 32; ++kt) {
            v16bf a = ld16(h_att, h_ao + kt * 64, 16);
            v16bf b = ld16(Wa, wbo + (unsigned)(DCTX * 2) + kt * 64, 16);
            acc = wmma_bf16(a, b, acc);
        }
        const int at = wave + 4 * i;
        float bv = ba[at * 16 + n];
#pragma unroll
        for (int v = 0; v < 8; ++v)
            att_s[(m0 + v) * DATT + at * 16 + n] = (__bf16)tanhf(acc[v] + bv);
    }
    __syncthreads();

    // ---- stage 3: GRU gates; wave owns h cols [wave*64, wave*64+64) ---------
    for (int ct = 0; ct < 4; ++ct) {
        const unsigned bih_o = wih_bo + ct * (unsigned)(16 * DATT * 2);
        const unsigned bhh_o = whh_bo + ct * (unsigned)(16 * LDH * 2);
        v8f gr = {}, gz = {}, gn = {}, hr = {}, hz = {}, hn = {};
        // unroll 2: two b128-pair A loads + six b128-pair B loads in flight per
        // clause -- enough latency hiding without spilling past the VGPR budget
        // (6 v8f accumulators stay live across both K loops).
#pragma unroll 2
        for (int kt = 0; kt < DATT / 32; ++kt) {
            v16bf a = ld16(att_s, att_ao + kt * 64, 16);
            gr = wmma_bf16(a, ld16(Wih, bih_o + kt * 64, 16), gr);
            gz = wmma_bf16(a, ld16(Wih, bih_o + 256u * DATT * 2 + kt * 64, 16), gz);
            gn = wmma_bf16(a, ld16(Wih, bih_o + 512u * DATT * 2 + kt * 64, 16), gn);
        }
#pragma unroll 2
        for (int kt = 0; kt < LDH / 32; ++kt) {
            v16bf a = ld16(h_gh, h_ao + kt * 64, 16);
            hr = wmma_bf16(a, ld16(Whh, bhh_o + kt * 64, 16), hr);
            hz = wmma_bf16(a, ld16(Whh, bhh_o + 256u * LDH * 2 + kt * 64, 16), hz);
            hn = wmma_bf16(a, ld16(Whh, bhh_o + 512u * LDH * 2 + kt * 64, 16), hn);
        }
        const int cbase = wave * 64 + ct * 16;
        const int col   = cbase + n;
        const float b_ir = bih[col], b_iz = bih[256 + col], b_in = bih[512 + col];
        const float b_hr = bhh[col], b_hz = bhh[256 + col], b_hn = bhh[512 + col];
#pragma unroll
        for (int v = 0; v < 8; ++v) {
            const int row = rbase + m0 + v;
            float rr = sigmoidf_((gr[v] + b_ir) + (hr[v] + b_hr));
            float zz = sigmoidf_((gz[v] + b_iz) + (hz[v] + b_hz));
            float nn = tanhf((gn[v] + b_in) + rr * (hn[v] + b_hn));
            float hp = FIRST ? h_prev[(size_t)row * 64 + (col & 63)]
                             : h_prev[(size_t)row * 256 + col];
            float hv = (1.0f - zz) * nn + zz * hp;
            h32_out[(size_t)row * 256 + col] = hv;
            __bf16 hb = (__bf16)hv;
            hbf_out[(size_t)row * 256 + col] = hb;
            hnew_s[(m0 + v) * 256 + col] = hb;
        }
    }
    __syncthreads();

    // ---- stage 4: out = h_new @ Wout.T + bout; wave -> out cols [16w,16w+16)
    {
        v8f acc = {};
#pragma unroll 4
        for (int kt = 0; kt < 8; ++kt) {
            v16bf a = ld16(hnew_s, hnew_ao + kt * 64, 16);
            v16bf b = ld16(WoutW, wout_bo + kt * 64, 16);
            acc = wmma_bf16(a, b, acc);
        }
        float bo = bout[wave * 16 + n];
#pragma unroll
        for (int v = 0; v < 8; ++v)
            outp[(size_t)(rbase + m0 + v) * 6400 + wave * 16 + n] = acc[v] + bo;
    }
}

// ---------------------------------------------------------------------------
// Host launcher
// ---------------------------------------------------------------------------
extern "C" void kernel_launch(void* const* d_in, const int* in_sizes, int n_in,
                              void* d_out, int out_size, void* d_ws, size_t ws_size,
                              hipStream_t stream) {
    (void)in_sizes; (void)n_in; (void)out_size; (void)ws_size;
    const float* fcn  = (const float*)d_in[0];   // 4096 x 64
    const float* enc  = (const float*)d_in[1];   // 4096 x 100 x 64
    const float* Wa1  = (const float*)d_in[2];   // 64 x 128
    const float* ba1  = (const float*)d_in[3];
    const float* Wa2  = (const float*)d_in[4];   // 256 x 512
    const float* ba2  = (const float*)d_in[5];
    const float* Wih1 = (const float*)d_in[6];   // 768 x 64
    const float* Whh1 = (const float*)d_in[7];   // 768 x 256
    const float* bih1 = (const float*)d_in[8];
    const float* bhh1 = (const float*)d_in[9];
    const float* Wih2 = (const float*)d_in[10];  // 768 x 256
    const float* Whh2 = (const float*)d_in[11];  // 768 x 256
    const float* bih2 = (const float*)d_in[12];
    const float* bhh2 = (const float*)d_in[13];
    const float* Wout = (const float*)d_in[14];  // 64 x 256
    const float* bout = (const float*)d_in[15];
    float* out = (float*)d_out;                  // 4096 x 100 x 64

    char* ws = (char*)d_ws;
    size_t off = 0;
    auto alloc = [&](size_t bytes) -> void* {
        void* p = ws + off;
        off = (off + bytes + 255) & ~(size_t)255;
        return p;
    };
    float*  h32[2]   = { (float*)alloc(4096u * 256 * 4), (float*)alloc(4096u * 256 * 4) };
    __bf16* hbf[2]   = { (__bf16*)alloc(4096u * 256 * 2), (__bf16*)alloc(4096u * 256 * 2) };
    __bf16* enc_bf   = (__bf16*)alloc((size_t)4096 * 100 * 64 * 2);
    __bf16* fcn_bf   = (__bf16*)alloc(4096u * 64 * 2);
    __bf16* wmat     = (__bf16*)alloc(256u * 64 * 2);
    __bf16* wa1_bf   = (__bf16*)alloc(64u * 128 * 2);
    __bf16* wa2_bf   = (__bf16*)alloc(256u * 512 * 2);
    __bf16* wih1_bf  = (__bf16*)alloc(768u * 64 * 2);
    __bf16* whh1f_bf = (__bf16*)alloc(768u * 64 * 2);
    __bf16* wih2_bf  = (__bf16*)alloc(768u * 256 * 2);
    __bf16* whh2_bf  = (__bf16*)alloc(768u * 256 * 2);
    __bf16* wout_bf  = (__bf16*)alloc(64u * 256 * 2);

    auto cvt = [&](const float* s, __bf16* d, int nel) {
        cvt_bf16_kernel<<<(nel + 255) / 256, 256, 0, stream>>>(s, d, nel);
    };
    cvt(enc,  enc_bf,  4096 * 100 * 64);
    cvt(fcn,  fcn_bf,  4096 * 64);
    cvt(Wa1,  wa1_bf,  64 * 128);
    cvt(Wa2,  wa2_bf,  256 * 512);
    cvt(Wih1, wih1_bf, 768 * 64);
    cvt(Wih2, wih2_bf, 768 * 256);
    cvt(Whh2, whh2_bf, 768 * 256);
    cvt(Wout, wout_bf, 64 * 256);
    fold_whh1_kernel<<<(768 * 64 + 255) / 256, 256, 0, stream>>>(Whh1, whh1f_bf);

    // ---- step 0 (FIRST): h0 = fcn (64-dim), produces hidden[0] -------------
    scores_softmax_kernel<<<4, 1024, 0, stream>>>(fcn, 64, enc, wmat);
    step_kernel<true><<<256, 128, 0, stream>>>(
        enc_bf, wmat, wa1_bf, ba1, wih1_bf, whh1f_bf, bih1, bhh1,
        fcn_bf, fcn_bf, fcn, h32[0], hbf[0], wout_bf, bout, out);

    // ---- steps 1..99 -------------------------------------------------------
    int cur = 0;
    for (int t = 1; t < 100; ++t) {
        scores_softmax_kernel<<<16, 1024, 0, stream>>>(
            h32[cur], 256, enc + (size_t)t * 64, wmat);
        step_kernel<false><<<256, 128, 0, stream>>>(
            enc_bf + (size_t)t * 64, wmat, wa2_bf, ba2, wih2_bf, whh2_bf,
            bih2, bhh2, hbf[cur], hbf[cur], h32[cur],
            h32[1 - cur], hbf[1 - cur], wout_bf, bout, out + (size_t)t * 64);
        cur = 1 - cur;
    }
}
